// AMM_38302518345900
// MI455X (gfx1250) — matrix operand order repaired and checked by reading, not compiled, not run on hardware
//
#include <hip/hip_runtime.h>

// MI455X / gfx1250, wave32. 1-NN via bf16 WMMA with fp32 hi/lo split.
// Phase 1: 256 blocks = 32 M-groups x 8 N-slices. Each block stages one
// prototype tile (hi+lo, 16KB) in LDS via async global->LDS copies and
// broadcasts it to 8 waves (each wave owns a 16-row M-tile, A resident in
// VGPRs) -> 8x less L2 traffic than per-wave streaming. Phase 2 reduces the
// 8 slice partials per row and gathers the matched f32 prototype rows.

typedef __attribute__((ext_vector_type(16))) __bf16 v16bf;
typedef __attribute__((ext_vector_type(8)))  float  v8f;
typedef __attribute__((ext_vector_type(4)))  float  v4f;
typedef int vsi4 __attribute__((vector_size(16)));   // matches builtin param

#define D_DIM   256
#define KCHUNKS 8          // 256 / 32 (bf16 WMMA K=32)
#define WAVES   8
#define SLICES  8
#define TILE_BYTES 8192    // one plane (hi or lo) of one nblk: 8 frags * 1KB
#define BUF_BYTES  16384   // hi + lo

// ---------------- async global->LDS helpers (CDNA5 path, guarded) ----------
__device__ __forceinline__ void copy16_g2s(const void* g, void* s) {
#if __has_builtin(__builtin_amdgcn_global_load_async_to_lds_b128)
  __builtin_amdgcn_global_load_async_to_lds_b128(
      (__attribute__((address_space(1))) vsi4*)g,
      (__attribute__((address_space(3))) vsi4*)s, 0, 0);
#else
  *(uint4*)s = *(const uint4*)g;           // fallback: via VGPRs + ds_store
#endif
}
__device__ __forceinline__ void wait_async() {
#if __has_builtin(__builtin_amdgcn_s_wait_asynccnt)
  __builtin_amdgcn_s_wait_asynccnt(0);
#elif __has_builtin(__builtin_amdgcn_global_load_async_to_lds_b128)
  asm volatile("s_wait_asynccnt 0x0" ::: "memory");
#endif
}

// ---------- pack x into WMMA A-fragment order (hi/lo bf16 planes) ----------
// A 16x32 bf16 layout (ISA 7.12.2): lane = M | (halfwave<<4),
//   half j -> K = j + 8*(j>>3) + 8*(lane>>4)
__global__ __launch_bounds__(256) void pack_A(const float* __restrict__ x,
                                              __bf16* __restrict__ Ahi,
                                              __bf16* __restrict__ Alo,
                                              int nfrag) {
  int gid  = blockIdx.x * 256 + threadIdx.x;
  int frag = gid >> 5, lane = gid & 31;
  if (frag >= nfrag) return;
  int mblk = frag >> 3, kc = frag & 7;
  int m = mblk * 16 + (lane & 15);
  v16bf vh, vl;
#pragma unroll
  for (int j = 0; j < 16; ++j) {
    int k = kc * 32 + j + ((j >> 3) << 3) + ((lane >> 4) << 3);
    float v = x[(size_t)m * D_DIM + k];
    __bf16 h = (__bf16)v;
    float  r = v - (float)h;
    vh[j] = h;
    vl[j] = (__bf16)r;
  }
  ((v16bf*)Ahi)[(size_t)frag * 32 + lane] = vh;
  ((v16bf*)Alo)[(size_t)frag * 32 + lane] = vl;
}

// ---------- pack prototypes into WMMA B-fragment order (hi/lo planes) ------
// B 32x16 bf16 layout: lane = N | (halfwave<<4), half j -> K = j + 8*(j>>3) + 8*(lane>>4)
__global__ __launch_bounds__(256) void pack_B(const float* __restrict__ p,
                                              __bf16* __restrict__ Bhi,
                                              __bf16* __restrict__ Blo,
                                              int nfrag) {
  int gid  = blockIdx.x * 256 + threadIdx.x;
  int frag = gid >> 5, lane = gid & 31;
  if (frag >= nfrag) return;
  int nblk = frag >> 3, kc = frag & 7;
  int n = nblk * 16 + (lane & 15);
  v16bf vh, vl;
#pragma unroll
  for (int j = 0; j < 16; ++j) {
    int k = kc * 32 + j + ((j >> 3) << 3) + ((lane >> 4) << 3);
    float v = p[(size_t)n * D_DIM + k];
    __bf16 h = (__bf16)v;
    float  r = v - (float)h;
    vh[j] = h;
    vl[j] = (__bf16)r;
  }
  ((v16bf*)Bhi)[(size_t)frag * 32 + lane] = vh;
  ((v16bf*)Blo)[(size_t)frag * 32 + lane] = vl;
}

// ---------- |p|^2, one wave per prototype row ----------
__global__ __launch_bounds__(256) void psq_kernel(const float* __restrict__ p,
                                                  float* __restrict__ p_sq,
                                                  int nrows) {
  int wave = threadIdx.x >> 5, lane = threadIdx.x & 31;
  int row = blockIdx.x * WAVES + wave;
  if (row >= nrows) return;
  const v4f* pr = (const v4f*)(p + (size_t)row * D_DIM + lane * 8);
  v4f a = pr[0], b = pr[1];
  float s = a.x * a.x + a.y * a.y + a.z * a.z + a.w * a.w +
            b.x * b.x + b.y * b.y + b.z * b.z + b.w * b.w;
#pragma unroll
  for (int off = 16; off >= 1; off >>= 1) s += __shfl_xor(s, off, 32);
  if (lane == 0) p_sq[row] = s;
}

// ---------- phase 1: LDS-shared B sweep over one N-slice -------------------
__global__ __launch_bounds__(256) void nn_sweep(
    const __bf16* __restrict__ Ahi, const __bf16* __restrict__ Alo,
    const __bf16* __restrict__ Bhi, const __bf16* __restrict__ Blo,
    const float* __restrict__ p_sq,
    float* __restrict__ part_val, int* __restrict__ part_idx,
    int per_slice) {
  __shared__ __attribute__((aligned(32))) char smem[2 * BUF_BYTES];

  const int slice  = blockIdx.x & (SLICES - 1);
  const int mgroup = blockIdx.x >> 3;
  const int wave   = threadIdx.x >> 5;
  const int lane   = threadIdx.x & 31;
  const int mblk   = mgroup * WAVES + wave;   // this wave's 16 query rows
  const int tid    = threadIdx.x;

  // A fragments resident in VGPRs (16 x 8 VGPRs = 128)
  const v16bf* Av_hi = (const v16bf*)Ahi;
  const v16bf* Av_lo = (const v16bf*)Alo;
  v16bf a_hi[KCHUNKS], a_lo[KCHUNKS];
#pragma unroll
  for (int kc = 0; kc < KCHUNKS; ++kc) {
    size_t idx = ((size_t)mblk * KCHUNKS + kc) * 32 + lane;
    a_hi[kc] = Av_hi[idx];
    a_lo[kc] = Av_lo[idx];
  }

  const char* gBhi = (const char*)Bhi;
  const char* gBlo = (const char*)Blo;
  const int   nbase = slice * per_slice;

  // cooperative tile copy: 256 threads x 4 x 16B = 16KB (hi 8K | lo 8K)
  auto load_tile = [&](int buf, int nblk) {
    char* s = smem + buf * BUF_BYTES;
    size_t gb = (size_t)nblk * TILE_BYTES;
#pragma unroll
    for (int part = 0; part < 2; ++part) {
      int off = tid * 16 + part * 4096;
      copy16_g2s(gBhi + gb + off, s + off);
      copy16_g2s(gBlo + gb + off, s + TILE_BYTES + off);
    }
  };

  float minv[8];
  int   mini[8];
#pragma unroll
  for (int r = 0; r < 8; ++r) { minv[r] = __builtin_inff(); mini[r] = 0; }

  load_tile(0, nbase);
  for (int i = 0; i < per_slice; ++i) {
    const int cur  = i & 1;
    const int nblk = nbase + i;
    wait_async();
    __syncthreads();                       // tile `cur` visible to all waves
    if (i + 1 < per_slice) load_tile(1 - cur, nblk + 1);

    const char* sb = smem + cur * BUF_BYTES;
    v8f acc = {};
#pragma unroll
    for (int kc = 0; kc < KCHUNKS; ++kc) {
      v16bf Bh = ((const v16bf*)(sb + kc * 1024))[lane];
      v16bf Bl = ((const v16bf*)(sb + TILE_BYTES + kc * 1024))[lane];
      // x.p = hi.hi + hi.lo + lo.hi (fp32-split bf16 emulation)
      acc = __builtin_amdgcn_wmma_f32_16x16x32_bf16(false, a_hi[kc], false, Bh,
                                                    (short)0, acc, false, false);
      acc = __builtin_amdgcn_wmma_f32_16x16x32_bf16(false, a_hi[kc], false, Bl,
                                                    (short)0, acc, false, false);
      acc = __builtin_amdgcn_wmma_f32_16x16x32_bf16(false, a_lo[kc], false, Bh,
                                                    (short)0, acc, false, false);
    }
    int   n  = nblk * 16 + (lane & 15);
    float ps = p_sq[n];
#pragma unroll
    for (int r = 0; r < 8; ++r) {
      float s = ps - 2.0f * acc[r];        // |x|^2 omitted: row-constant
      if (s < minv[r]) { minv[r] = s; mini[r] = n; }
    }
    __syncthreads();                       // all waves done reading `cur`
  }

  // reduce over the 16 lanes holding the same 8 rows (half-wave groups)
#pragma unroll
  for (int off = 8; off >= 1; off >>= 1) {
#pragma unroll
    for (int r = 0; r < 8; ++r) {
      float ov = __shfl_xor(minv[r], off, 32);
      int   oi = __shfl_xor(mini[r], off, 32);
      if (ov < minv[r] || (ov == minv[r] && oi < mini[r])) {
        minv[r] = ov; mini[r] = oi;
      }
    }
  }
  if (lane == 0) {
#pragma unroll
    for (int r = 0; r < 8; ++r) {
      int row = mblk * 16 + r;
      part_val[(size_t)row * SLICES + slice] = minv[r];
      part_idx[(size_t)row * SLICES + slice] = mini[r];
    }
  }
  if (lane == 16) {
#pragma unroll
    for (int r = 0; r < 8; ++r) {
      int row = mblk * 16 + 8 + r;
      part_val[(size_t)row * SLICES + slice] = minv[r];
      part_idx[(size_t)row * SLICES + slice] = mini[r];
    }
  }
}

// ---------- phase 2: slice reduction + gather ------------------------------
__global__ __launch_bounds__(256) void nn_final(
    const float* __restrict__ part_val, const int* __restrict__ part_idx,
    const float* __restrict__ protos, float* __restrict__ out) {
  __shared__ int final_idx[16];
  int t = threadIdx.x;
  if (t < 16) {
    size_t row = (size_t)blockIdx.x * 16 + t;
    float bv = part_val[row * SLICES];
    int   bi = part_idx[row * SLICES];
#pragma unroll
    for (int s = 1; s < SLICES; ++s) {
      float v = part_val[row * SLICES + s];
      int   i = part_idx[row * SLICES + s];
      if (v < bv || (v == bv && i < bi)) { bv = v; bi = i; }
    }
    final_idx[t] = bi;
  }
  __syncthreads();
  int row = t >> 4, col = (t & 15) * 16;
  const v4f* src = (const v4f*)(protos + (size_t)final_idx[row] * D_DIM + col);
  v4f* dst = (v4f*)(out + ((size_t)blockIdx.x * 16 + row) * D_DIM + col);
#pragma unroll
  for (int i = 0; i < 4; ++i) dst[i] = src[i];
}

extern "C" void kernel_launch(void* const* d_in, const int* in_sizes, int n_in,
                              void* d_out, int out_size, void* d_ws, size_t ws_size,
                              hipStream_t stream) {
  (void)n_in; (void)out_size; (void)ws_size;
  const float* x      = (const float*)d_in[0];
  const float* protos = (const float*)d_in[1];
  float* out = (float*)d_out;

  const int Brows = in_sizes[0] / D_DIM;   // 4096
  const int Nrows = in_sizes[1] / D_DIM;   // 32768

  // ws: Ahi 2MB | Alo 2MB | Bhi 16MB | Blo 16MB | p_sq 128K | pval 128K | pidx 128K
  size_t a_elems = (size_t)Brows * D_DIM;
  size_t b_elems = (size_t)Nrows * D_DIM;
  __bf16* Ahi = (__bf16*)d_ws;
  __bf16* Alo = Ahi + a_elems;
  __bf16* Bhi = Alo + a_elems;
  __bf16* Blo = Bhi + b_elems;
  float*  p_sq = (float*)(Blo + b_elems);
  float*  pval = p_sq + Nrows;
  int*    pidx = (int*)(pval + (size_t)Brows * SLICES);

  int nfragA = (Brows / 16) * KCHUNKS;
  int nfragB = (Nrows / 16) * KCHUNKS;
  int per_slice = (Nrows / 16) / SLICES;   // 256 nblks per slice
  int mgroups   = Brows / (16 * WAVES);    // 32

  pack_A<<<(nfragA * 32 + 255) / 256, 256, 0, stream>>>(x, Ahi, Alo, nfragA);
  pack_B<<<(nfragB * 32 + 255) / 256, 256, 0, stream>>>(protos, Bhi, Blo, nfragB);
  psq_kernel<<<(Nrows + WAVES - 1) / WAVES, 256, 0, stream>>>(protos, p_sq, Nrows);
  nn_sweep<<<mgroups * SLICES, 256, 0, stream>>>(Ahi, Alo, Bhi, Blo, p_sq,
                                                 pval, pidx, per_slice);
  nn_final<<<Brows / 16, 256, 0, stream>>>(pval, pidx, protos, out);
}
